// EquivariantScalar_viaTP_84748294684816
// MI455X (gfx1250) — matrix operand order repaired and checked by reading, not compiled, not run on hardware
//
#include <hip/hip_runtime.h>
#include <hip/hip_bf16.h>
#include <stdint.h>

typedef __attribute__((ext_vector_type(16))) __bf16 v16bf;
typedef __attribute__((ext_vector_type(8)))  float  v8f;

#define AATTR 32
#define VEC_STRIDE 480                // 128 + 3*64 + 5*32
#define W1_PACK_ELEMS (128*8*32*16)   // 524288 bf16 (1 MB)
#define W2_PACK_ELEMS (128*2*32*16)   // 131072 bf16 (256 KB)
#define WAVES 8

// ---------------------------------------------------------------------------
// gfx1250 async global->LDS copies (ASYNCcnt-tracked). GVS addressing:
// vdst = LDS byte offset, vaddr = 32-bit byte offset, saddr = 64-bit base.
// ---------------------------------------------------------------------------
__device__ __forceinline__ void async_copy_b128(unsigned lds_off, unsigned goff,
                                                const void* base) {
    asm volatile("global_load_async_to_lds_b128 %0, %1, %2"
                 :: "v"(lds_off), "v"(goff), "s"(base) : "memory");
}
__device__ __forceinline__ void async_copy_b64(unsigned lds_off, unsigned goff,
                                               const void* base) {
    asm volatile("global_load_async_to_lds_b64 %0, %1, %2"
                 :: "v"(lds_off), "v"(goff), "s"(base) : "memory");
}
#define WAIT_ASYNC(n) asm volatile("s_wait_asynccnt " #n ::: "memory")

__device__ __forceinline__ unsigned lds_off_of(const void* p) {
    return (unsigned)(uintptr_t)p;   // LDS offset lives in the low 32 bits
}

union BFrag { uint4 q[2]; v16bf v; };

// ---------------------------------------------------------------------------
// Kernel 1: pack W1s/64 and W2/64 (f32) into bf16 WMMA B-fragment layout.
// B (32x16, bf16), k order chosen here: element e of lane L holds K = e + (L>=16 ? 16 : 0),
// column N = L%16. Fragment order: [u][wtile][lane][e].
// ---------------------------------------------------------------------------
__global__ void pack_weights_kernel(const float* __restrict__ W1s,
                                    const float* __restrict__ W2,
                                    __hip_bfloat16* __restrict__ p1,
                                    __hip_bfloat16* __restrict__ p2) {
    int tid = blockIdx.x * blockDim.x + threadIdx.x;
    const float scale = 1.0f / 64.0f;             // inv = 1/sqrt(128*32)
    if (tid < W1_PACK_ELEMS) {
        int e    = tid & 15;
        int lane = (tid >> 4) & 31;
        int t    = (tid >> 9) & 7;
        int u    = tid >> 12;
        int v    = e + ((lane >= 16) ? 16 : 0);
        int w    = t * 16 + (lane & 15);
        p1[tid] = __hip_bfloat16(W1s[(u * 32 + v) * 128 + w] * scale);
    } else if (tid < W1_PACK_ELEMS + W2_PACK_ELEMS) {
        int q    = tid - W1_PACK_ELEMS;
        int e    = q & 15;
        int lane = (q >> 4) & 31;
        int t    = (q >> 9) & 1;
        int u    = q >> 10;
        int v    = e + ((lane >= 16) ? 16 : 0);
        int w2   = t * 16 + (lane & 15);
        p2[q] = __hip_bfloat16(W2[(u * 32 + v) * 32 + w2] * scale);
    }
}

// ---------------------------------------------------------------------------
// Kernel 2: fused network. 8 waves/block, one 16-node tile per wave.
// Weights for each 32-wide K-chunk are staged ONCE per block into LDS with
// double-buffered async-to-LDS copies and shared by all 8 waves (8x less L2
// weight traffic). Each K-chunk (fixed u) is rank-1:
// A[m,j] = s[m,u]*attr[m,j], so the A-fragment is the cached attr fragment
// scaled by one scalar per chunk. B-fragments are double-buffered in VGPRs
// so each WMMA overlaps the next tile's ds_load_b128 pair.
// ---------------------------------------------------------------------------
__global__ __launch_bounds__(256) void
fused_equiv_kernel(const float* __restrict__ node_vec,
                   const float* __restrict__ attr,
                   const __hip_bfloat16* __restrict__ p1,
                   const __hip_bfloat16* __restrict__ p2,
                   const float* __restrict__ b1s,
                   const float* __restrict__ b2v,
                   const float* __restrict__ W3,
                   const float* __restrict__ b3,
                   const float* __restrict__ W4,
                   const float* __restrict__ b4,
                   float* __restrict__ out) {
    __shared__ float s_lds[WAVES][16 * 128];          // s tile, reused for s_act
    __shared__ float h_lds[WAVES][16 * 32];
    __shared__ alignas(16) unsigned char wstage[2 * 8192];  // double-buffered weights

    const int tid  = threadIdx.x;
    const int lane = tid & 31;
    const int wave = tid >> 5;
    const int node_base = (blockIdx.x * WAVES + wave) * 16;
    const int col  = lane & 15;   // C/D column, also A row (node within tile)
    const int half = lane >> 4;

    // ---- stage s tile (16 nodes x 128) into LDS, coalesced ----
    for (int rr = 0; rr < 16; ++rr)
        for (int c = lane; c < 128; c += 32)
            s_lds[wave][rr * 128 + c] =
                node_vec[(size_t)(node_base + rr) * VEC_STRIDE + c];

    // ---- attr A-fragment (constant over all K chunks) ----
    // A 16x32 bf16, k order used here (matches B pack): lanes<16: K=e (0..15)
    //                      lanes>=16: K=e+16 (16..31)
    float attrf[16];
#pragma unroll
    for (int e = 0; e < 16; ++e) {
        int v = e + half * 16;
        attrf[e] = attr[(node_base + col) * AATTR + v];
    }

    const v8f vzero = {0.f, 0.f, 0.f, 0.f, 0.f, 0.f, 0.f, 0.f};
    v8f acc[8];
#pragma unroll
    for (int t = 0; t < 8; ++t) acc[t] = vzero;

    const unsigned wbase = lds_off_of(&wstage[0]);

    // ---- GEMM1: s_mid_raw[16x128], K staged per-chunk via async-to-LDS ----
    // Chunk u occupies 8 KB: [t(8)][lane(32)][16 bf16]. 256 threads stage
    // 32 B each (two b128 async copies).
    {
        unsigned d0 = wbase + tid * 32;              // prefetch chunk 0
        async_copy_b128(d0,      tid * 32,      p1);
        async_copy_b128(d0 + 16, tid * 32 + 16, p1);
    }
#pragma unroll 1
    for (int u = 0; u < 128; ++u) {
        const int cur = u & 1;
        if (u + 1 < 128) {                           // prefetch next chunk
            unsigned dst = wbase + ((u + 1) & 1) * 8192 + tid * 32;
            unsigned src = (unsigned)(u + 1) * 8192 + tid * 32;
            async_copy_b128(dst,      src,      p1);
            async_copy_b128(dst + 16, src + 16, p1);
            WAIT_ASYNC(2);                           // chunk u landed (in-order)
        } else {
            WAIT_ASYNC(0);
        }
        __syncthreads();                             // all waves' stage visible

        float sval = s_lds[wave][col * 128 + u];
        v16bf a;
#pragma unroll
        for (int e = 0; e < 16; ++e) a[e] = (__bf16)(sval * attrf[e]);

        const uint4* wb = (const uint4*)(wstage + cur * 8192);
        // B-fragment software pipeline: issue loads for tile t+1 before
        // the WMMA of tile t so ds latency overlaps the matrix pipe.
        BFrag f[2];
        f[0].q[0] = wb[lane * 2];
        f[0].q[1] = wb[lane * 2 + 1];
#pragma unroll
        for (int t = 0; t < 8; ++t) {
            if (t < 7) {
                f[(t + 1) & 1].q[0] = wb[(t + 1) * 64 + lane * 2];
                f[(t + 1) & 1].q[1] = wb[(t + 1) * 64 + lane * 2 + 1];
            }
            acc[t] = __builtin_amdgcn_wmma_f32_16x16x32_bf16(
                false, a, false, f[t & 1].v, (short)0, acc[t], false, false);
        }
        __syncthreads();                             // buffer free for reuse
    }

    // ---- kick off GEMM2 chunk-0 stage (2 KB: one b64 per thread) ----
    async_copy_b64(wbase + tid * 8, tid * 8, p2);

    // ---- epilogue 1: + b1s, silu, write s_act in place over s tile ----
    // C/D layout: lane holds col N=lane%16; VGPR r -> row M = r + half*8
#pragma unroll
    for (int t = 0; t < 8; ++t) {
        int w = t * 16 + col;
        float bias = b1s[w];
#pragma unroll
        for (int r = 0; r < 8; ++r) {
            int m = r + half * 8;
            float x = acc[t][r] + bias;
            s_lds[wave][m * 128 + w] = x / (1.0f + __expf(-x));
        }
    }

    // ---- GEMM2: h_raw[16x32] with s_act as the scaling stream ----
    v8f acc2[2];
    acc2[0] = vzero; acc2[1] = vzero;
#pragma unroll 1
    for (int u = 0; u < 128; ++u) {
        const int cur = u & 1;
        if (u + 1 < 128) {
            unsigned dst = wbase + ((u + 1) & 1) * 2048 + tid * 8;
            unsigned src = (unsigned)(u + 1) * 2048 + tid * 8;
            async_copy_b64(dst, src, p2);
            WAIT_ASYNC(1);
        } else {
            WAIT_ASYNC(0);
        }
        __syncthreads();

        float sval = s_lds[wave][col * 128 + u];
        v16bf a;
#pragma unroll
        for (int e = 0; e < 16; ++e) a[e] = (__bf16)(sval * attrf[e]);

        const uint4* wb = (const uint4*)(wstage + cur * 2048);
        BFrag f0, f1;                                // load both tiles up front
        f0.q[0] = wb[lane * 2];
        f0.q[1] = wb[lane * 2 + 1];
        f1.q[0] = wb[64 + lane * 2];
        f1.q[1] = wb[64 + lane * 2 + 1];
        acc2[0] = __builtin_amdgcn_wmma_f32_16x16x32_bf16(
            false, a, false, f0.v, (short)0, acc2[0], false, false);
        acc2[1] = __builtin_amdgcn_wmma_f32_16x16x32_bf16(
            false, a, false, f1.v, (short)0, acc2[1], false, false);
        __syncthreads();
    }

    // ---- epilogue 2: + b2 -> h_lds ----
#pragma unroll
    for (int t = 0; t < 2; ++t) {
        int w2 = t * 16 + col;
        float bias = b2v[w2];
#pragma unroll
        for (int r = 0; r < 8; ++r) {
            int m = r + half * 8;
            h_lds[wave][m * 32 + w2] = acc2[t][r] + bias;
        }
    }

    // ---- final tiny stage: silu(h @ W3/sqrt(A) + b3) @ W4/sqrt(A) + b4 ----
    if (lane < 16) {
        const float rsq = 0.17677669529663687f;  // 1/sqrt(32)
        float hv[32];
#pragma unroll
        for (int i = 0; i < 32; ++i) hv[i] = h_lds[wave][lane * 32 + i];
        float e = b4[0];
        for (int j = 0; j < 32; ++j) {
            float z = 0.0f;
#pragma unroll
            for (int i = 0; i < 32; ++i) z += hv[i] * W3[i * 32 + j];
            z = z * rsq + b3[j];
            float y = z / (1.0f + __expf(-z));
            e += y * W4[j] * rsq;
        }
        out[node_base + lane] = e;
    }
}

extern "C" void kernel_launch(void* const* d_in, const int* in_sizes, int n_in,
                              void* d_out, int out_size, void* d_ws, size_t ws_size,
                              hipStream_t stream) {
    (void)in_sizes; (void)n_in; (void)out_size; (void)ws_size;
    const float* node_vec = (const float*)d_in[0];
    const float* attr     = (const float*)d_in[1];
    const float* W1s      = (const float*)d_in[2];
    const float* b1s      = (const float*)d_in[3];
    // d_in[4..7] = W1g, b1g, W1v1, W1v2 : dead code in the reference
    const float* W2       = (const float*)d_in[8];
    const float* b2v      = (const float*)d_in[9];
    const float* W3       = (const float*)d_in[10];
    const float* b3       = (const float*)d_in[11];
    const float* W4       = (const float*)d_in[12];
    const float* b4       = (const float*)d_in[13];
    float* out = (float*)d_out;

    __hip_bfloat16* p1 = (__hip_bfloat16*)d_ws;
    __hip_bfloat16* p2 = p1 + W1_PACK_ELEMS;

    const int pack_total = W1_PACK_ELEMS + W2_PACK_ELEMS;
    pack_weights_kernel<<<(pack_total + 255) / 256, 256, 0, stream>>>(W1s, W2, p1, p2);

    // 8192 nodes / (16 per wave * 8 waves) = 64 blocks of 256 threads
    fused_equiv_kernel<<<64, 256, 0, stream>>>(node_vec, attr, p1, p2,
                                               b1s, b2v, W3, b3, W4, b4, out);
}
